// AttentionTransformer_7902739825141
// MI455X (gfx1250) — matrix operand order repaired and checked by reading, not compiled, not run on hardware
//
#include <hip/hip_runtime.h>

typedef __bf16 bf16;
typedef __attribute__((ext_vector_type(16))) __bf16 v16bf;
typedef __attribute__((ext_vector_type(8)))  __bf16 bf8;
typedef __attribute__((ext_vector_type(4)))  __bf16 bf4;
typedef __attribute__((ext_vector_type(8)))  float  v8f;
typedef __attribute__((ext_vector_type(4)))  float  f4;

#define T_SEQ 2048
#define CDIM  1024
#define NH    16
#define DH    64
#define BSZ   2
#define MROWS (BSZ * T_SEQ)
#define NEGV  -4294967295.0f

// native truncation: lets the backend use CDNA5 bf16 cvt hardware instead of
// a manual integer RNE sequence (4+ VALU/elem previously)
static __device__ __forceinline__ bf16 f2bf(float f) { return (bf16)f; }

static __device__ __forceinline__ v8f wmma_bf16(v16bf a, v16bf b, v8f c) {
  // (neg_a, A, neg_b, B, c_mod, C, reuse_a, reuse_b)
  return __builtin_amdgcn_wmma_f32_16x16x32_bf16(false, a, false, b, (short)0, c,
                                                 false, false);
}

static __device__ __forceinline__ v16bf cat16(bf8 lo, bf8 hi) {
  return __builtin_shufflevector(lo, hi, 0, 1, 2, 3, 4, 5, 6, 7,
                                 8, 9, 10, 11, 12, 13, 14, 15);
}

// ---------------------------------------------------------------------------
// Kernel 1: out = relu(X @ W + bias), bf16 output. X:[4096,1024] W:[1024,1024]
// WG tile 128x128, 8 waves of 32x64, k-step 32.
// ---------------------------------------------------------------------------
__global__ __launch_bounds__(256) void qkv_kernel(const float* __restrict__ X,
                                                  const float* __restrict__ W,
                                                  const float* __restrict__ bias,
                                                  bf16* __restrict__ out) {
  __shared__ bf16 As[128 * 48];  // A tile, row-major [m][k], stride 48
  __shared__ bf16 Bs[128 * 48];  // B tile transposed [n][k], stride 48

  const int tid  = threadIdx.x;
  const int lane = tid & 31;
  const int wave = tid >> 5;
  const int h    = lane >> 4;
  const int ln   = lane & 15;
  const int wm   = wave & 3;   // 4 waves along M
  const int wn   = wave >> 2;  // 2 waves along N
  const int m0   = blockIdx.y * 128;
  const int n0   = blockIdx.x * 128;

  v8f acc[2][4];
#pragma unroll
  for (int i = 0; i < 2; ++i)
#pragma unroll
    for (int j = 0; j < 4; ++j) acc[i][j] = (v8f){0.f, 0.f, 0.f, 0.f, 0.f, 0.f, 0.f, 0.f};

  for (int k0 = 0; k0 < CDIM; k0 += 32) {
    __syncthreads();
    // stage A: 128x32 fp32 -> bf16
#pragma unroll
    for (int r = 0; r < 4; ++r) {
      int row = r * 32 + (tid >> 3);
      int c4  = (tid & 7) * 4;
      f4 v = *(const f4*)&X[(size_t)(m0 + row) * CDIM + k0 + c4];
      if (k0 + 32 < CDIM)
        __builtin_prefetch(&X[(size_t)(m0 + row) * CDIM + k0 + 32 + c4], 0, 3);
      *(bf4*)&As[row * 48 + c4] = __builtin_convertvector(v, bf4);
    }
    // stage B transposed: 32x128 fp32 -> bf16 [n][k]
#pragma unroll
    for (int r = 0; r < 4; ++r) {
      int kr  = r * 8 + wave;
      int nc4 = lane * 4;
      f4 v = *(const f4*)&W[(size_t)(k0 + kr) * CDIM + n0 + nc4];
      if (k0 + 32 < CDIM)
        __builtin_prefetch(&W[(size_t)(k0 + 32 + kr) * CDIM + n0 + nc4], 0, 3);
      Bs[(nc4 + 0) * 48 + kr] = f2bf(v.x);
      Bs[(nc4 + 1) * 48 + kr] = f2bf(v.y);
      Bs[(nc4 + 2) * 48 + kr] = f2bf(v.z);
      Bs[(nc4 + 3) * 48 + kr] = f2bf(v.w);
    }
    __syncthreads();

    v16bf afr[2], bfr[4];
#pragma unroll
    for (int i = 0; i < 2; ++i) {
      int row = wm * 32 + 16 * i + ln;
      bf8 lo = *(const bf8*)&As[row * 48 + 8 * h];        // K = 8h .. 8h+7
      bf8 hi = *(const bf8*)&As[row * 48 + 16 + 8 * h];   // K = 16+8h ..
      afr[i] = cat16(lo, hi);
    }
#pragma unroll
    for (int j = 0; j < 4; ++j) {
      int col = wn * 64 + 16 * j + ln;
      bf8 lo = *(const bf8*)&Bs[col * 48 + 8 * h];        // K = 8h .. 8h+7
      bf8 hi = *(const bf8*)&Bs[col * 48 + 16 + 8 * h];   // K = 16+8h ..
      bfr[j] = cat16(lo, hi);
    }
#pragma unroll
    for (int i = 0; i < 2; ++i)
#pragma unroll
      for (int j = 0; j < 4; ++j) acc[i][j] = wmma_bf16(afr[i], bfr[j], acc[i][j]);
  }

  // epilogue: bias + relu -> bf16
#pragma unroll
  for (int i = 0; i < 2; ++i) {
    int rbase = m0 + wm * 32 + 16 * i + 8 * h;
#pragma unroll
    for (int j = 0; j < 4; ++j) {
      int col = n0 + wn * 64 + 16 * j + ln;
      float bb = bias[col];
#pragma unroll
      for (int vv = 0; vv < 8; ++vv) {
        float val = acc[i][j][vv] + bb;
        val = fmaxf(val, 0.0f);
        out[(size_t)(rbase + vv) * CDIM + col] = f2bf(val);
      }
    }
  }
}

// ---------------------------------------------------------------------------
// Kernel 2: row mask = sign(|sum(x, -1)|)  (0.0 or 1.0), one wave per row
// ---------------------------------------------------------------------------
__global__ __launch_bounds__(256) void rowmask_kernel(const float* __restrict__ X,
                                                      float* __restrict__ mask) {
  int row  = blockIdx.x * 8 + (threadIdx.x >> 5);
  int lane = threadIdx.x & 31;
  float s = 0.f;
  for (int c = lane; c < CDIM; c += 32) s += X[(size_t)row * CDIM + c];
#pragma unroll
  for (int off = 16; off > 0; off >>= 1) s += __shfl_xor(s, off, 32);
  if (lane == 0) mask[row] = (s != 0.0f) ? 1.0f : 0.0f;
}

// ---------------------------------------------------------------------------
// Kernel 3: flash attention per (q-tile=64, head, batch). 4 waves, 16 q-rows
// per wave. Causal + key mask with exact NEG constant, online softmax in fp32.
// ---------------------------------------------------------------------------
__global__ __launch_bounds__(128) void attn_kernel(const bf16* __restrict__ Qb,
                                                   const bf16* __restrict__ Kb,
                                                   const bf16* __restrict__ Vb,
                                                   const float* __restrict__ mask,
                                                   float* __restrict__ O) {
  __shared__ bf16 vt[64 * 72];       // V^T tile: [dh][key], stride 72
  __shared__ bf16 pb[4][16 * 72];    // per-wave P tile: [qrow][key], stride 72

  const int tid  = threadIdx.x;
  const int lane = tid & 31;
  const int wave = tid >> 5;
  const int h    = lane >> 4;
  const int ln   = lane & 15;
  const int qt   = blockIdx.x;
  const int head = blockIdx.y;
  const int b    = blockIdx.z;
  const size_t rowbase = (size_t)b * T_SEQ;
  const int coff  = head * DH;
  const int qrow0 = qt * 64 + wave * 16;

  // Q A-fragments (16 rows x 64 dh = two K-32 fragments), loaded once
  v16bf qa0, qa1;
  {
    const bf16* qp = &Qb[(rowbase + qrow0 + ln) * CDIM + coff];
    qa0 = cat16(*(const bf8*)&qp[8 * h], *(const bf8*)&qp[16 + 8 * h]);
    qa1 = cat16(*(const bf8*)&qp[32 + 8 * h], *(const bf8*)&qp[48 + 8 * h]);
  }

  float mi[8], li[8];
  v8f oacc[4];
#pragma unroll
  for (int vv = 0; vv < 8; ++vv) { mi[vv] = -3.0e38f; li[vv] = 0.0f; }
#pragma unroll
  for (int ct = 0; ct < 4; ++ct) oacc[ct] = (v8f){0.f, 0.f, 0.f, 0.f, 0.f, 0.f, 0.f, 0.f};

  for (int kt = 0; kt <= qt; ++kt) {
    const int ktb = kt * 64;
    __syncthreads();
    // stage V^T cooperatively; prefetch next V tile while we are at it
#pragma unroll
    for (int r = 0; r < 4; ++r) {
      int key = r * 16 + (tid >> 3);
      int d0  = (tid & 7) * 8;
      bf8 vv8 = *(const bf8*)&Vb[(rowbase + ktb + key) * CDIM + coff + d0];
      if (kt < qt)
        __builtin_prefetch(&Vb[(rowbase + ktb + 64 + key) * CDIM + coff + d0], 0, 3);
#pragma unroll
      for (int j = 0; j < 8; ++j) vt[(d0 + j) * 72 + key] = vv8[j];
    }
    __syncthreads();

    // S = Q K^T * (1/sqrt(dh)), masked
    float pv[4][8];
#pragma unroll
    for (int ct = 0; ct < 4; ++ct) {
      int kg = ktb + ct * 16 + ln;
      const bf16* kp = &Kb[(rowbase + kg) * CDIM + coff];
      if (kt < qt)
        __builtin_prefetch(&Kb[(rowbase + kg + 64) * CDIM + coff], 0, 3);
      v16bf kb0 = cat16(*(const bf8*)&kp[8 * h], *(const bf8*)&kp[16 + 8 * h]);
      v16bf kb1 = cat16(*(const bf8*)&kp[32 + 8 * h], *(const bf8*)&kp[48 + 8 * h]);
      v8f s = (v8f){0.f, 0.f, 0.f, 0.f, 0.f, 0.f, 0.f, 0.f};
      s = wmma_bf16(qa0, kb0, s);
      s = wmma_bf16(qa1, kb1, s);
      float km = mask[rowbase + kg];
#pragma unroll
      for (int vv = 0; vv < 8; ++vv) {
        int qg = qrow0 + vv + 8 * h;
        float sv = s[vv] * 0.125f;
        pv[ct][vv] = (km != 0.0f && kg <= qg) ? sv : NEGV;
      }
    }

    // online softmax (rows live in 16-lane halves: xor 1..8)
#pragma unroll
    for (int vv = 0; vv < 8; ++vv) {
      float m = fmaxf(fmaxf(pv[0][vv], pv[1][vv]), fmaxf(pv[2][vv], pv[3][vv]));
#pragma unroll
      for (int off = 1; off <= 8; off <<= 1) m = fmaxf(m, __shfl_xor(m, off, 32));
      float mn    = fmaxf(mi[vv], m);
      float alpha = __expf(mi[vv] - mn);
      mi[vv] = mn;
      li[vv] *= alpha;
#pragma unroll
      for (int ct = 0; ct < 4; ++ct) oacc[ct][vv] *= alpha;
      float rs = 0.f;
#pragma unroll
      for (int ct = 0; ct < 4; ++ct) {
        float p = __expf(pv[ct][vv] - mn);
        pb[wave][(vv + 8 * h) * 72 + ct * 16 + ln] = f2bf(p);
        rs += p;
      }
#pragma unroll
      for (int off = 1; off <= 8; off <<= 1) rs += __shfl_xor(rs, off, 32);
      li[vv] += rs;
    }

    // O += P @ V   (P re-shaped via wave-private LDS into A-fragments)
    const bf16* pw = &pb[wave][0];
    v16bf pa0 = cat16(*(const bf8*)&pw[ln * 72 + 8 * h],
                      *(const bf8*)&pw[ln * 72 + 16 + 8 * h]);
    v16bf pa1 = cat16(*(const bf8*)&pw[ln * 72 + 32 + 8 * h],
                      *(const bf8*)&pw[ln * 72 + 48 + 8 * h]);
#pragma unroll
    for (int ct = 0; ct < 4; ++ct) {
      int col = ct * 16 + ln;
      v16bf vb0 = cat16(*(const bf8*)&vt[col * 72 + 8 * h],
                        *(const bf8*)&vt[col * 72 + 16 + 8 * h]);
      v16bf vb1 = cat16(*(const bf8*)&vt[col * 72 + 32 + 8 * h],
                        *(const bf8*)&vt[col * 72 + 48 + 8 * h]);
      oacc[ct] = wmma_bf16(pa0, vb0, oacc[ct]);
      oacc[ct] = wmma_bf16(pa1, vb1, oacc[ct]);
    }
  }

  // normalize and write O (fp32, head-interleaved layout [B*T][U])
#pragma unroll
  for (int vv = 0; vv < 8; ++vv) {
    float inv = 1.0f / li[vv];
    size_t r = rowbase + qrow0 + vv + 8 * h;
#pragma unroll
    for (int ct = 0; ct < 4; ++ct)
      O[r * CDIM + coff + ct * 16 + ln] = oacc[ct][vv] * inv;
  }
}

// ---------------------------------------------------------------------------
// Kernel 4: out = LN(O*qmask + x) * gamma + beta   (biased var, eps inside sqrt)
// ---------------------------------------------------------------------------
__global__ __launch_bounds__(256) void ln_kernel(const float* __restrict__ O,
                                                 const float* __restrict__ X,
                                                 const float* __restrict__ mask,
                                                 const float* __restrict__ gamma,
                                                 const float* __restrict__ beta,
                                                 float* __restrict__ out) {
  __shared__ float ws1[8], ws2[8];
  __shared__ float smean, srstd;
  const int row  = blockIdx.x;
  const int tid  = threadIdx.x;
  const int lane = tid & 31;
  const int wave = tid >> 5;
  float qm = mask[row];
  float vals[4];
  float s = 0.f, s2 = 0.f;
#pragma unroll
  for (int i = 0; i < 4; ++i) {
    int c = tid + i * 256;
    float v = O[(size_t)row * CDIM + c] * qm + X[(size_t)row * CDIM + c];
    vals[i] = v;
    s += v;
    s2 += v * v;
  }
#pragma unroll
  for (int off = 16; off > 0; off >>= 1) {
    s += __shfl_xor(s, off, 32);
    s2 += __shfl_xor(s2, off, 32);
  }
  if (lane == 0) { ws1[wave] = s; ws2[wave] = s2; }
  __syncthreads();
  if (tid == 0) {
    float S = 0.f, S2 = 0.f;
    for (int j = 0; j < 8; ++j) { S += ws1[j]; S2 += ws2[j]; }
    float mean = S * (1.0f / CDIM);
    float var  = S2 * (1.0f / CDIM) - mean * mean;
    smean = mean;
    srstd = rsqrtf(var + 1e-8f);
  }
  __syncthreads();
  float mean = smean, rstd = srstd;
#pragma unroll
  for (int i = 0; i < 4; ++i) {
    int c = tid + i * 256;
    out[(size_t)row * CDIM + c] = gamma[c] * ((vals[i] - mean) * rstd) + beta[c];
  }
}

// ---------------------------------------------------------------------------
extern "C" void kernel_launch(void* const* d_in, const int* in_sizes, int n_in,
                              void* d_out, int out_size, void* d_ws, size_t ws_size,
                              hipStream_t stream) {
  (void)in_sizes; (void)n_in; (void)out_size; (void)ws_size;
  const float* x     = (const float*)d_in[0];
  const float* Wq    = (const float*)d_in[1];
  const float* bq    = (const float*)d_in[2];
  const float* Wk    = (const float*)d_in[3];
  const float* bk    = (const float*)d_in[4];
  const float* Wv    = (const float*)d_in[5];
  const float* bv    = (const float*)d_in[6];
  const float* gamma = (const float*)d_in[7];
  const float* beta  = (const float*)d_in[8];
  float* out = (float*)d_out;

  char* ws = (char*)d_ws;
  const size_t szBf = (size_t)MROWS * CDIM * sizeof(bf16);
  const size_t szF  = (size_t)MROWS * CDIM * sizeof(float);
  bf16*  Qb  = (bf16*)(ws);
  bf16*  Kb  = (bf16*)(ws + szBf);
  bf16*  Vb  = (bf16*)(ws + 2 * szBf);
  float* O   = (float*)(ws + 3 * szBf);
  float* msk = (float*)(ws + 3 * szBf + szF);

  dim3 gq(CDIM / 128, MROWS / 128);
  qkv_kernel<<<gq, 256, 0, stream>>>(x, Wq, bq, Qb);
  qkv_kernel<<<gq, 256, 0, stream>>>(x, Wk, bk, Kb);
  qkv_kernel<<<gq, 256, 0, stream>>>(x, Wv, bv, Vb);
  rowmask_kernel<<<MROWS / 8, 256, 0, stream>>>(x, msk);
  attn_kernel<<<dim3(T_SEQ / 64, NH, BSZ), 128, 0, stream>>>(Qb, Kb, Vb, msk, O);
  ln_kernel<<<MROWS, 256, 0, stream>>>(O, x, msk, gamma, beta, out);
}
